// ScaledDotProduct_38611755991512
// MI455X (gfx1250) — matrix operand order repaired and checked by reading, not compiled, not run on hardware
//
#include <hip/hip_runtime.h>
#include <math.h>

// CDNA5 / gfx1250 flash-attention:
//   out = softmax(QK^T * 1/sqrt(D) masked + bias) @ V
// One wave32 owns a 16-row Q tile; KV streamed in 32-wide chunks.
// QK^T and P@V run on v_wmma_f32_16x16x32_f16 (f16 in, f32 accumulate).

typedef __attribute__((ext_vector_type(16))) _Float16 v16h;
typedef __attribute__((ext_vector_type(8)))  float    v8f;

#define B_ 32
#define S_ 2048
#define D_ 64
#define WAVES_PER_BLOCK 8

__global__ __launch_bounds__(256, 2) void fa_wmma_kernel(
    const float* __restrict__ q, const float* __restrict__ k,
    const float* __restrict__ v, const unsigned char* __restrict__ mask,
    const float* __restrict__ bias, float* __restrict__ out) {

  // per-wave 16x32 f16 staging buffer for the P (C-layout -> A-layout) transpose
  __shared__ _Float16 lds[WAVES_PER_BLOCK * 16 * 32];

  const int lane = threadIdx.x & 31;
  const int wave = threadIdx.x >> 5;
  const int gw   = blockIdx.x * WAVES_PER_BLOCK + wave;
  const int qtiles = S_ / 16;            // 128
  const int b  = gw / qtiles;
  const int q0 = (gw % qtiles) * 16;

  _Float16* pbuf = &lds[wave * 16 * 32];

  const int m  = lane & 15;   // A-layout row / B-layout column (per lane)
  const int kg = lane >> 4;   // half-wave selector

  // ---- Q A-fragments (two 16x32 f16 frags covering K=0..63), pre-scaled ----
  // A layout: lane holds row m; element i holds K = kg*16 + i
  const float scale = 0.125f; // 1/sqrt(64)
  const float* qrow = q + ((size_t)b * S_ + q0 + m) * D_;
  v16h a0, a1;
#pragma unroll
  for (int i = 0; i < 16; ++i) {
    const int kk = (kg << 4) + i;
    a0[i] = (_Float16)(qrow[kk]      * scale);
    a1[i] = (_Float16)(qrow[kk + 32] * scale);
  }

  // online-softmax state: this lane's 8 rows are m = r + 8*kg (C/D layout)
  float mrow[8], lrow[8];
  v8f acc[4] = {v8f{}, v8f{}, v8f{}, v8f{}};   // 16x64 fp32 accumulator
#pragma unroll
  for (int r = 0; r < 8; ++r) { mrow[r] = -3.0e38f; lrow[r] = 0.0f; }

  const size_t mbase   = (size_t)b * S_ * S_;
  const float* kb_ptr  = k + (size_t)b * S_ * D_;
  const float* vb_ptr  = v + (size_t)b * S_ * D_;

  for (int kv0 = 0; kv0 < S_; kv0 += 32) {
    // CDNA5 speculative prefetch of the next K chunk (global_prefetch_b8)
    if (kv0 + 32 < S_)
      __builtin_prefetch(kb_ptr + (size_t)(kv0 + 32 + lane) * D_, 0, 0);

    // ---- scores S[16 x 32]: two 16x16 WMMA tiles, K chained 0..31, 32..63 ----
    v8f st[2];
#pragma unroll
    for (int nt = 0; nt < 2; ++nt) {
      // B layout: lane holds column (kv) = kv0+nt*16+m; element i holds
      // K-dim = kg*16 + i  (first step) / 32 + kg*16 + i (second step).
      const float* krow = kb_ptr + (size_t)(kv0 + nt * 16 + m) * D_ + (kg << 4);
      v16h b0, b1;
#pragma unroll
      for (int i = 0; i < 16; ++i) {
        b0[i] = (_Float16)krow[i];
        b1[i] = (_Float16)krow[i + 32];
      }
      v8f s = {};
      s = __builtin_amdgcn_wmma_f32_16x16x32_f16(false, a0, false, b0,
                                                 (short)0, s, false, false);
      s = __builtin_amdgcn_wmma_f32_16x16x32_f16(false, a1, false, b1,
                                                 (short)0, s, false, false);
      st[nt] = s;
    }

    // ---- mask + bias in C layout (lane = column, VGPR r = row r+8*kg) ----
#pragma unroll
    for (int nt = 0; nt < 2; ++nt) {
      const int col = kv0 + nt * 16 + m;
#pragma unroll
      for (int r = 0; r < 8; ++r) {
        const int qg = q0 + r + (kg << 3);
        const bool keep = mask[mbase + (size_t)qg * S_ + col] != 0;
        st[nt][r] = keep ? (st[nt][r] + bias[(size_t)qg * S_ + col]) : -3.0e38f;
      }
    }

    // ---- online softmax: row reductions are 16-lane butterflies ----
#pragma unroll
    for (int r = 0; r < 8; ++r) {
      float vmax = fmaxf(st[0][r], st[1][r]);
      vmax = fmaxf(vmax, __shfl_xor(vmax, 1));
      vmax = fmaxf(vmax, __shfl_xor(vmax, 2));
      vmax = fmaxf(vmax, __shfl_xor(vmax, 4));
      vmax = fmaxf(vmax, __shfl_xor(vmax, 8));
      const float mnew  = fmaxf(mrow[r], vmax);
      const float alpha = __expf(mrow[r] - mnew);
      const float p0 = __expf(st[0][r] - mnew);
      const float p1 = __expf(st[1][r] - mnew);
      st[0][r] = p0;
      st[1][r] = p1;
      float psum = p0 + p1;
      psum += __shfl_xor(psum, 1);
      psum += __shfl_xor(psum, 2);
      psum += __shfl_xor(psum, 4);
      psum += __shfl_xor(psum, 8);
      lrow[r] = lrow[r] * alpha + psum;
      mrow[r] = mnew;
      acc[0][r] *= alpha; acc[1][r] *= alpha;
      acc[2][r] *= alpha; acc[3][r] *= alpha;
    }

    // ---- transpose P: C layout -> A layout via per-wave LDS tile ----
#pragma unroll
    for (int nt = 0; nt < 2; ++nt) {
      const int col = nt * 16 + m;
#pragma unroll
      for (int r = 0; r < 8; ++r)
        pbuf[(r + (kg << 3)) * 32 + col] = (_Float16)st[nt][r];
    }
    // LDS ops are in-order per wave; explicit split-counter wait for clarity
    asm volatile("s_wait_dscnt 0" ::: "memory");
    v16h pa;
#pragma unroll
    for (int i = 0; i < 16; ++i) {
      const int col = (kg << 4) + i;
      pa[i] = pbuf[m * 32 + col];
    }

    // ---- acc[16 x 64] += P[16 x 32] @ V[32 x 64] : 4 WMMAs ----
#pragma unroll
    for (int dt = 0; dt < 4; ++dt) {
      // B layout: lane holds column d = dt*16+m; element i holds kv row
      // kv0 + kg*16 + i  -> stride-D column gather (coalesced across lanes)
      const float* vcol = vb_ptr + (size_t)(kv0 + (kg << 4)) * D_ + dt * 16 + m;
      v16h bv;
#pragma unroll
      for (int i = 0; i < 16; ++i) bv[i] = (_Float16)vcol[(size_t)i * D_];
      acc[dt] = __builtin_amdgcn_wmma_f32_16x16x32_f16(false, pa, false, bv,
                                                       (short)0, acc[dt],
                                                       false, false);
    }
  }

  // ---- epilogue: normalize and store (C layout -> row-major) ----
  float* orow = out + ((size_t)b * S_ + q0) * D_;
#pragma unroll
  for (int r = 0; r < 8; ++r) {
    const int row = r + (kg << 3);
    const float inv = (lrow[r] > 0.0f) ? (1.0f / lrow[r]) : 0.0f;
#pragma unroll
    for (int dt = 0; dt < 4; ++dt)
      orow[(size_t)row * D_ + dt * 16 + m] = acc[dt][r] * inv;
  }
}

extern "C" void kernel_launch(void* const* d_in, const int* in_sizes, int n_in,
                              void* d_out, int out_size, void* d_ws, size_t ws_size,
                              hipStream_t stream) {
  (void)in_sizes; (void)n_in; (void)out_size; (void)d_ws; (void)ws_size;
  const float*         q    = (const float*)d_in[0];
  const float*         k    = (const float*)d_in[1];
  const float*         v    = (const float*)d_in[2];
  const unsigned char* mask = (const unsigned char*)d_in[3];  // jnp bool -> 1 byte
  const float*         bias = (const float*)d_in[4];
  float* out = (float*)d_out;

  const int total_waves = B_ * (S_ / 16);               // 4096 Q tiles
  dim3 grid(total_waves / WAVES_PER_BLOCK);             // 512 blocks
  dim3 block(WAVES_PER_BLOCK * 32);                     // 8 waves (wave32)
  fa_wmma_kernel<<<grid, block, 0, stream>>>(q, k, v, mask, bias, out);
}
